// Part_Graph_51539607552415
// MI455X (gfx1250) — compile-verified, matrix operand order checked by reading, not run on hardware
//
#include <hip/hip_runtime.h>
#include <cstddef>

typedef __attribute__((ext_vector_type(16))) _Float16 v16h;
typedef __attribute__((ext_vector_type(8)))  float    v8f;

namespace {
constexpr int Bn = 2, Hh = 96, Ww = 96, HW = Hh * Ww;   // 9216
constexpr int INC = 256, HID = 10, Pp = 6, Ee = 12;
constexpr int C2 = 2 * HID;          // 20
constexpr int OC = 27;               // offset-conv output channels
constexpr int Kc = C2 * 9;           // 180  (conv GEMM K)
constexpr int KcPad = 192;           // padded to 6 * 32
constexpr int Nn = Bn * HW;          // 18432 (GEMM N)
constexpr int NT = Nn / 16;          // 1152 N-tiles
constexpr float EPS = 1e-5f;

__device__ __forceinline__ float sigmoidf(float x) { return 1.f / (1.f + __expf(-x)); }
}

// -------------------- small utility kernels --------------------

__global__ void k_zero(float* __restrict__ p, int n) {
  int i = blockIdx.x * blockDim.x + threadIdx.x;
  if (i < n) p[i] = 0.f;
}

// xcat[e,b,c,hw] = xp[(c<10?src:dst)[e], b, c%10, hw]
__global__ void k_xcat(const float* __restrict__ xp, const int* __restrict__ esrc,
                       const int* __restrict__ edst, float* __restrict__ xcat) {
  int idx = blockIdx.x * blockDim.x + threadIdx.x;
  if (idx >= Ee * Bn * C2 * HW) return;
  int hw = idx % HW; int t = idx / HW;
  int c = t % C2; t /= C2;
  int b = t % Bn; int e = t / Bn;
  int part = (c < HID) ? esrc[e] : edst[e];
  int cc   = (c < HID) ? c : c - HID;
  xcat[idx] = xp[(((size_t)part * Bn + b) * HID + cc) * HW + hw];
}

// -------------------- WMMA implicit-GEMM 3x3 conv (20 -> 27) --------------------
// out[e,b,m,hw] = bias[e,m] + sum_{c<20,kh,kw} in[e,b,c,(h+kh-1)(w+kw-1)] * wgt[e,m,c,kh,kw]
// M=27 (pad 32: 2 m-tiles), K=180 (pad 192), N=18432 (1152 n-tiles). One wave per tile.
__global__ __launch_bounds__(128) void k_conv_wmma(const float* __restrict__ xin,
                                                   const float* __restrict__ wgt,
                                                   const float* __restrict__ bias,
                                                   float* __restrict__ out) {
  const int lane = threadIdx.x & 31;
  const int wave = blockIdx.x * (blockDim.x >> 5) + (threadIdx.x >> 5);
  if (wave >= Ee * 2 * NT) return;                    // wave-uniform exit
  const int nt = wave % NT;
  int tq = wave / NT;
  const int mt = tq & 1;
  const int e  = tq >> 1;
  const int hi8 = (lane >> 4) * 8;                   // K sub-block select per half-wave
  const int l16 = lane & 15;
  const int n  = nt * 16 + l16;
  const int b  = n / HW; const int hw = n % HW;
  const int h  = hw / Ww; const int w = hw % Ww;
  const float* xpb = xin + ((size_t)e * Bn + b) * C2 * HW;
  const float* wp  = wgt + (size_t)e * OC * Kc;
  const int m_a = mt * 16 + l16;

  v8f acc = {};
#pragma unroll
  for (int k0 = 0; k0 < KcPad; k0 += 32) {
    v16h av, bv;
#pragma unroll
    for (int j = 0; j < 16; ++j) {
      int kk = k0 + ((j < 8) ? j : (j + 8)) + hi8;   // 16-bit A/B fragment K mapping
      float a = 0.f, bbv = 0.f;
      if (kk < Kc) {
        if (m_a < OC) a = wp[m_a * Kc + kk];
        int c = kk / 9, tap = kk % 9;
        int ih = h + tap / 3 - 1, iw = w + tap % 3 - 1;
        if (ih >= 0 && ih < Hh && iw >= 0 && iw < Ww) bbv = xpb[c * HW + ih * Ww + iw];
      }
      av[j] = (_Float16)a;
      bv[j] = (_Float16)bbv;
    }
    acc = __builtin_amdgcn_wmma_f32_16x16x32_f16(false, av, false, bv, (short)0, acc,
                                                 false, false);
  }
  float* op = out + ((size_t)e * Bn + b) * OC * HW + hw;
#pragma unroll
  for (int r = 0; r < 8; ++r) {
    int m = mt * 16 + r + hi8;
    if (m < OC) op[m * HW] = acc[r] + bias[e * OC + m];
  }
}

// -------------------- fused deformable sampling + channel einsum + BN stats --------------------
template <int COUT>
__global__ __launch_bounds__(256) void k_dcn(const float* __restrict__ xin,   // [E,B,20,HW]
                                             const float* __restrict__ om,    // [E,B,27,HW]
                                             const float* __restrict__ wgt,   // [E,COUT,20,9]
                                             float* __restrict__ out,         // [E,B,COUT,HW]
                                             float* __restrict__ stats) {     // [E*COUT*2]
  __shared__ float sw[COUT * C2 * 9];
  __shared__ float ssum[COUT], ssq[COUT];
  const int e_blk = (int)(((size_t)blockIdx.x * blockDim.x) / (Bn * HW));   // uniform per block
  for (int i = threadIdx.x; i < COUT * C2 * 9; i += blockDim.x)
    sw[i] = wgt[(size_t)e_blk * COUT * C2 * 9 + i];
  if (threadIdx.x < COUT) { ssum[threadIdx.x] = 0.f; ssq[threadIdx.x] = 0.f; }
  __syncthreads();

  const int idx = blockIdx.x * blockDim.x + threadIdx.x;   // grid sized exactly E*B*HW
  const int hw = idx % HW; int t = idx / HW;
  const int b = t % Bn; const int e = t / Bn;
  const int h = hw / Ww, w = hw % Ww;
  const float* omp = om + ((size_t)e * Bn + b) * OC * HW + hw;
  const float* xpb = xin + ((size_t)e * Bn + b) * C2 * HW;

  float acc[COUT];
#pragma unroll
  for (int o = 0; o < COUT; ++o) acc[o] = 0.f;

  for (int k = 0; k < 9; ++k) {
    float dy = omp[(2 * k) * HW];
    float dx = omp[(2 * k + 1) * HW];
    float mk = sigmoidf(omp[(18 + k) * HW]);
    float ys = (float)(h + k / 3 - 1) + dy;
    float xs = (float)(w + k % 3 - 1) + dx;
    float y0 = floorf(ys), x0 = floorf(xs);
    float wy = ys - y0, wx = xs - x0;
    int iy = (int)y0, ix = (int)x0;
    bool vy0 = (iy >= 0) && (iy < Hh), vy1 = (iy + 1 >= 0) && (iy + 1 < Hh);
    bool vx0 = (ix >= 0) && (ix < Ww), vx1 = (ix + 1 >= 0) && (ix + 1 < Ww);
    int cy0 = min(max(iy, 0), Hh - 1), cy1 = min(max(iy + 1, 0), Hh - 1);
    int cx0 = min(max(ix, 0), Ww - 1), cx1 = min(max(ix + 1, 0), Ww - 1);
    float w00 = (vy0 && vx0) ? (1.f - wy) * (1.f - wx) * mk : 0.f;
    float w01 = (vy0 && vx1) ? (1.f - wy) * wx * mk : 0.f;
    float w10 = (vy1 && vx0) ? wy * (1.f - wx) * mk : 0.f;
    float w11 = (vy1 && vx1) ? wy * wx * mk : 0.f;
    int i00 = cy0 * Ww + cx0, i01 = cy0 * Ww + cx1;
    int i10 = cy1 * Ww + cx0, i11 = cy1 * Ww + cx1;
    for (int c = 0; c < C2; ++c) {
      const float* xc = xpb + c * HW;
      float v = w00 * xc[i00] + w01 * xc[i01] + w10 * xc[i10] + w11 * xc[i11];
#pragma unroll
      for (int o = 0; o < COUT; ++o) acc[o] += v * sw[(o * C2 + c) * 9 + k];
    }
  }

  float* op = out + ((size_t)e * Bn + b) * COUT * HW + hw;
#pragma unroll
  for (int o = 0; o < COUT; ++o) {
    op[o * HW] = acc[o];
    atomicAdd(&ssum[o], acc[o]);
    atomicAdd(&ssq[o], acc[o] * acc[o]);
  }
  __syncthreads();
  if (threadIdx.x < COUT) {
    atomicAdd(&stats[(e_blk * COUT + threadIdx.x) * 2 + 0], ssum[threadIdx.x]);
    atomicAdd(&stats[(e_blk * COUT + threadIdx.x) * 2 + 1], ssq[threadIdx.x]);
  }
}

// -------------------- per-(edge,channel) BN + ReLU (in place) --------------------
template <int C>
__global__ void k_bn_relu(float* __restrict__ x, const float* __restrict__ st,
                          const float* __restrict__ g, const float* __restrict__ beta) {
  int idx = blockIdx.x * blockDim.x + threadIdx.x;
  if (idx >= Ee * Bn * C * HW) return;
  int t = idx / HW;
  int c = t % C; t /= C; t /= Bn;   // t now = e
  int m = t * C + c;
  const float Nf = (float)(Bn * HW);
  float mu = st[2 * m] / Nf;
  float var = st[2 * m + 1] / Nf - mu * mu;
  float v = g[m] * (x[idx] - mu) * rsqrtf(var + EPS) + beta[m];
  x[idx] = fmaxf(v, 0.f);
}

// -------------------- edge message: sigmoid dots --------------------
__global__ void k_msg(const float* __restrict__ h2, const float* __restrict__ xp,
                      const float* __restrict__ out_w, const float* __restrict__ out_b,
                      const float* __restrict__ aatt_w, const float* __restrict__ aatt_b,
                      const int* __restrict__ esrc, float* __restrict__ msg) {
  int idx = blockIdx.x * blockDim.x + threadIdx.x;
  if (idx >= Ee * Bn * HW) return;
  int hw = idx % HW; int t = idx / HW;
  int b = t % Bn; int e = t / Bn;
  float s1 = out_b[e];
  const float* hp = h2 + ((size_t)e * Bn + b) * HID * HW + hw;
  for (int o = 0; o < HID; ++o) s1 += hp[o * HW] * out_w[e * HID + o];
  float s2 = aatt_b[e];
  const float* ap = xp + ((size_t)esrc[e] * Bn + b) * HID * HW + hw;
  for (int c = 0; c < HID; ++c) s2 += ap[c * HW] * aatt_w[e * HID + c];
  msg[idx] = sigmoidf(s1) * (1.f - sigmoidf(s2));
}

// -------------------- gate = 1 + decomp + self_att + segment_sum(msg) --------------------
__global__ void k_gate(const float* __restrict__ xp, const float* __restrict__ xh,
                       const float* __restrict__ att_w, const float* __restrict__ att_b,
                       const float* __restrict__ du_w, const float* __restrict__ du_b,
                       const float* __restrict__ dl_w, const float* __restrict__ dl_b,
                       const float* __restrict__ msg, const int* __restrict__ edst,
                       float* __restrict__ gate) {
  int idx = blockIdx.x * blockDim.x + threadIdx.x;
  if (idx >= Pp * Bn * HW) return;
  int hw = idx % HW; int t = idx / HW;
  int b = t % Bn; int p = t / Bn;
  // per-part self attention (1x1 over HID)
  float sa = att_b[p];
  const float* xpp_ = xp + ((size_t)p * Bn + b) * HID * HW + hw;
  for (int c = 0; c < HID; ++c) sa += xpp_[c * HW] * att_w[p * HID + c];
  sa = sigmoidf(sa);
  // half-body decomposition gate (parts 0..3 upper, 4..5 lower)
  int half = (p < 4) ? 0 : 1;
  const float* hww = half ? dl_w : du_w;
  float ds = half ? dl_b[0] : du_b[0];
  const float* hp = xh + ((size_t)half * Bn + b) * HID * HW + hw;
  for (int c = 0; c < HID; ++c) ds += hp[c * HW] * hww[c];
  float dec = sigmoidf(ds);
  // segment sum of edge messages by destination
  float xs = 0.f;
  for (int e = 0; e < Ee; ++e)
    if (edst[e] == p) xs += msg[((size_t)e * Bn + b) * HW + hw];
  gate[idx] = 1.f + dec + sa + xs;
}

// -------------------- WMMA update GEMM fused with gate scaling + BN stats --------------------
// t[m=p*10+o, n=(b,hw)] = sum_c upd_w[p,o,c] * p_fea[b,c,hw];  y = t * gate[p,b,hw]
__global__ __launch_bounds__(128) void k_update_wmma(const float* __restrict__ p_fea,
                                                     const float* __restrict__ upd_w,
                                                     const float* __restrict__ gate,
                                                     float* __restrict__ y,
                                                     float* __restrict__ stats) {
  const int lane = threadIdx.x & 31;
  const int wave = blockIdx.x * (blockDim.x >> 5) + (threadIdx.x >> 5);
  if (wave >= 4 * NT) return;                         // M padded 60 -> 64 (4 m-tiles)
  const int nt = wave % NT; const int mt = wave / NT;
  const int hi8 = (lane >> 4) * 8; const int l16 = lane & 15;
  const int n = nt * 16 + l16;
  const int b = n / HW; const int hw = n % HW;
  const int m_a = mt * 16 + l16;

  v8f acc = {};
#pragma unroll
  for (int k0 = 0; k0 < INC; k0 += 32) {
    if (k0 + 32 < INC)
      __builtin_prefetch(&p_fea[((size_t)b * INC + k0 + 32 + hi8) * HW + hw], 0, 0);
    v16h av, bv;
#pragma unroll
    for (int j = 0; j < 16; ++j) {
      int kk = k0 + ((j < 8) ? j : (j + 8)) + hi8;
      av[j] = (_Float16)((m_a < Pp * HID) ? upd_w[m_a * INC + kk] : 0.f);
      bv[j] = (_Float16)p_fea[((size_t)b * INC + kk) * HW + hw];
    }
    acc = __builtin_amdgcn_wmma_f32_16x16x32_f16(false, av, false, bv, (short)0, acc,
                                                 false, false);
  }
#pragma unroll
  for (int r = 0; r < 8; ++r) {
    int m = mt * 16 + r + hi8;
    if (m < Pp * HID) {
      int p = m / HID, o = m % HID;
      float g = gate[((size_t)p * Bn + b) * HW + hw];
      float v = acc[r] * g;
      y[(((size_t)p * Bn + b) * HID + o) * HW + hw] = v;
      atomicAdd(&stats[2 * m + 0], v);
      atomicAdd(&stats[2 * m + 1], v * v);
    }
  }
}

// -------------------- final per-(part,channel) BN + ReLU --------------------
__global__ void k_bn5(const float* __restrict__ y, const float* __restrict__ st,
                      const float* __restrict__ g, const float* __restrict__ beta,
                      float* __restrict__ out) {
  int idx = blockIdx.x * blockDim.x + threadIdx.x;
  if (idx >= Pp * Bn * HID * HW) return;
  int t = idx / HW;
  int o = t % HID; t /= HID; t /= Bn;   // t now = p
  int m = t * HID + o;
  const float Nf = (float)(Bn * HW);
  float mu = st[2 * m] / Nf;
  float var = st[2 * m + 1] / Nf - mu * mu;
  float v = g[m] * (y[idx] - mu) * rsqrtf(var + EPS) + beta[m];
  out[idx] = fmaxf(v, 0.f);
}

// -------------------- host launcher --------------------
extern "C" void kernel_launch(void* const* d_in, const int* in_sizes, int n_in,
                              void* d_out, int out_size, void* d_ws, size_t ws_size,
                              hipStream_t stream) {
  (void)in_sizes; (void)n_in; (void)out_size; (void)ws_size;
  const float* p_fea  = (const float*)d_in[0];
  const float* xp     = (const float*)d_in[1];
  const float* xh     = (const float*)d_in[2];
  const float* att_w  = (const float*)d_in[3];
  const float* att_b  = (const float*)d_in[4];
  const float* du_w   = (const float*)d_in[5];
  const float* du_b   = (const float*)d_in[6];
  const float* dl_w   = (const float*)d_in[7];
  const float* dl_b   = (const float*)d_in[8];
  const float* upd_w  = (const float*)d_in[9];
  const float* upd_g  = (const float*)d_in[10];
  const float* upd_b  = (const float*)d_in[11];
  const float* off1_w = (const float*)d_in[12];
  const float* off1_b = (const float*)d_in[13];
  const float* dcn1_w = (const float*)d_in[14];
  const float* bn1_g  = (const float*)d_in[15];
  const float* bn1_b  = (const float*)d_in[16];
  const float* off2_w = (const float*)d_in[17];
  const float* off2_b = (const float*)d_in[18];
  const float* dcn2_w = (const float*)d_in[19];
  const float* bn2_g  = (const float*)d_in[20];
  const float* bn2_b  = (const float*)d_in[21];
  const float* out_w  = (const float*)d_in[22];
  const float* out_b  = (const float*)d_in[23];
  const float* aatt_w = (const float*)d_in[24];
  const float* aatt_b = (const float*)d_in[25];
  const int*   esrc   = (const int*)d_in[26];
  const int*   edst   = (const int*)d_in[27];
  float* out = (float*)d_out;

  // scratch layout (floats)
  float* ws   = (float*)d_ws;
  float* XCAT = ws;                              // E*B*20*HW = 4,423,680
  float* OM   = XCAT + (size_t)Ee * Bn * C2 * HW;   // E*B*27*HW = 5,971,968
  float* H1   = OM   + (size_t)Ee * Bn * OC * HW;   // E*B*20*HW
  float* H2   = H1   + (size_t)Ee * Bn * C2 * HW;   // E*B*10*HW = 2,211,840
  float* TB   = H2   + (size_t)Ee * Bn * HID * HW;  // P*B*10*HW = 1,105,920
  float* GATE = TB   + (size_t)Pp * Bn * HID * HW;  // P*B*HW
  float* MSG  = GATE + (size_t)Pp * Bn * HW;        // E*B*HW
  float* ST1  = MSG  + (size_t)Ee * Bn * HW;        // E*20*2 = 480
  float* ST2  = ST1 + Ee * C2 * 2;                  // E*10*2 = 240
  float* ST5  = ST2 + Ee * HID * 2;                 // 60*2   = 120

  const int nstat = Ee * C2 * 2 + Ee * HID * 2 + Pp * HID * 2;
  k_zero<<<(nstat + 255) / 256, 256, 0, stream>>>(ST1, nstat);

  // 1) concatenate endpoint features per edge
  k_xcat<<<(Ee * Bn * C2 * HW + 255) / 256, 256, 0, stream>>>(xp, esrc, edst, XCAT);

  // 2) offset conv 1 (WMMA implicit GEMM), then deformable conv 1 + BN + ReLU
  k_conv_wmma<<<(Ee * 2 * NT) / 4, 128, 0, stream>>>(XCAT, off1_w, off1_b, OM);
  k_dcn<C2><<<(Ee * Bn * HW) / 256, 256, 0, stream>>>(XCAT, OM, dcn1_w, H1, ST1);
  k_bn_relu<C2><<<(Ee * Bn * C2 * HW + 255) / 256, 256, 0, stream>>>(H1, ST1, bn1_g, bn1_b);

  // 3) offset conv 2, deformable conv 2 + BN + ReLU
  k_conv_wmma<<<(Ee * 2 * NT) / 4, 128, 0, stream>>>(H1, off2_w, off2_b, OM);
  k_dcn<HID><<<(Ee * Bn * HW) / 256, 256, 0, stream>>>(H1, OM, dcn2_w, H2, ST2);
  k_bn_relu<HID><<<(Ee * Bn * HID * HW + 255) / 256, 256, 0, stream>>>(H2, ST2, bn2_g, bn2_b);

  // 4) edge messages, gate
  k_msg<<<(Ee * Bn * HW + 255) / 256, 256, 0, stream>>>(H2, xp, out_w, out_b, aatt_w,
                                                        aatt_b, esrc, MSG);
  k_gate<<<(Pp * Bn * HW + 255) / 256, 256, 0, stream>>>(xp, xh, att_w, att_b, du_w, du_b,
                                                         dl_w, dl_b, MSG, edst, GATE);

  // 5) update GEMM (WMMA) fused with gate scale + BN stats, then final BN + ReLU
  k_update_wmma<<<(4 * NT) / 4, 128, 0, stream>>>(p_fea, upd_w, GATE, TB, ST5);
  k_bn5<<<(Pp * Bn * HID * HW + 255) / 256, 256, 0, stream>>>(TB, ST5, upd_g, upd_b, out);
}
